// PGLU_b_13340168421984
// MI455X (gfx1250) — matrix operand order repaired and checked by reading, not compiled, not run on hardware
//
#include <hip/hip_runtime.h>
#include <math.h>

typedef __attribute__((ext_vector_type(16))) __bf16 v16bf;
typedef __attribute__((ext_vector_type(8)))  __bf16 v8bf;
typedef __attribute__((ext_vector_type(4)))  __bf16 v4bf;
typedef __attribute__((ext_vector_type(8)))  float  v8f;

#define BATCH 1024
#define ISZ   2048
#define HSZ   2048
#define KSZ   (ISZ + HSZ)   // 4096
#define BM    128
#define BN    256
#define BK    32
#define KT    (KSZ / BK)    // 128 k-steps

// ---------------- fragment helpers (CDNA5 16-bit WMMA lane layouts) -------------
// A 16x32 bf16: lanes 0-15 hold M=lane, K in {0-7, 16-23}; lanes 16-31 hold
// M=lane-16, K in {8-15, 24-31}.  LDS A tile stored row-major [BM][BK].
__device__ __forceinline__ v16bf load_a_frag(const __bf16* As, int fragM, int lane) {
    const int m     = fragM + (lane & 15);
    const int khalf = (lane >> 4) * 8;
    v8bf lo = *reinterpret_cast<const v8bf*>(As + m * BK + khalf);        // K 0-7 / 8-15
    v8bf hi = *reinterpret_cast<const v8bf*>(As + m * BK + 16 + khalf);   // K 16-23 / 24-31
    return __builtin_shufflevector(lo, hi, 0,1,2,3,4,5,6,7,8,9,10,11,12,13,14,15);
}

// B 32x16 bf16: lanes 0-15 hold col=lane, K in {0-7, 16-23}; lanes 16-31 hold col=lane-16,
// K in {8-15, 24-31}.  LDS B tile stored as [BN][BK] (output-col major, contiguous K).
__device__ __forceinline__ v16bf load_b_frag(const __bf16* Bs, int fragN, int lane) {
    const int n     = fragN + (lane & 15);
    const int kbase = (lane >> 4) * 8;
    v8bf lo = *reinterpret_cast<const v8bf*>(Bs + n * BK + kbase);
    v8bf hi = *reinterpret_cast<const v8bf*>(Bs + n * BK + kbase + 16);
    return __builtin_shufflevector(lo, hi, 0,1,2,3,4,5,6,7,8,9,10,11,12,13,14,15);
}

__device__ __forceinline__ v4bf cvt4(float4 f) {
    v4bf r;
    r[0] = (__bf16)f.x; r[1] = (__bf16)f.y; r[2] = (__bf16)f.z; r[3] = (__bf16)f.w;
    return r;
}

__device__ __forceinline__ float sigmoidf_(float v) {
    return 1.0f / (1.0f + __expf(-v));
}

// Stage one (BM x BK) A-tile and (BN x BK) B-tile, fp32 -> bf16, into LDS buffer.
// asrc has row stride 2048 (x / h_prev / rh all [.,2048]); W has row stride KSZ.
__device__ __forceinline__ void stage_tiles(
    __bf16* As, __bf16* Bs, const float* __restrict__ asrc,
    const float* __restrict__ W, int blockM, int blockN, int k0,
    int tid, bool prefetch_next)
{
#pragma unroll
    for (int i = 0; i < 4; ++i) {
        const int q   = tid + i * 256;
        const int row = q >> 3;
        const int c4  = (q & 7) * 4;
        float4 v = *reinterpret_cast<const float4*>(asrc + (size_t)(blockM + row) * 2048 + c4);
        *reinterpret_cast<v4bf*>(&As[row * BK + c4]) = cvt4(v);
    }
#pragma unroll
    for (int i = 0; i < 8; ++i) {
        const int q   = tid + i * 256;
        const int row = q >> 3;                   // output column n
        const int c4  = (q & 7) * 4;
        const float* wp = W + (size_t)(blockN + row) * KSZ + k0 + c4;
        float4 v = *reinterpret_cast<const float4*>(wp);
        if (prefetch_next) __builtin_prefetch(wp + BK, 0, 0);
        *reinterpret_cast<v4bf*>(&Bs[row * BK + c4]) = cvt4(v);
    }
}

// 16 WMMAs (4x4 tile of 16x16) for one K-step out of an LDS buffer.
__device__ __forceinline__ void mma_step(
    const __bf16* As, const __bf16* Bs, v8f acc[4][4], int wm, int wn, int lane)
{
    v16bf af[4], bfv[4];
#pragma unroll
    for (int i = 0; i < 4; ++i) af[i]  = load_a_frag(As, wm * 64 + i * 16, lane);
#pragma unroll
    for (int j = 0; j < 4; ++j) bfv[j] = load_b_frag(Bs, wn * 64 + j * 16, lane);
#pragma unroll
    for (int i = 0; i < 4; ++i)
#pragma unroll
        for (int j = 0; j < 4; ++j)
            acc[i][j] = __builtin_amdgcn_wmma_f32_16x16x32_bf16(
                false, af[i], false, bfv[j], (short)0, acc[i][j], false, false);
}

// =====================  kernel 1: r & z GEMMs (fused)  =====================
// blockIdx.z = 0 -> rh = sigmoid(A@W_r^T + b_r) * h_prev   (only use of r)
// blockIdx.z = 1 -> z  = sigmoid(A@W_z^T + b_z)
__global__ __launch_bounds__(256) void pglu_gemm_rz(
    const float* __restrict__ x,   const float* __restrict__ h_prev,
    const float* __restrict__ W_r, const float* __restrict__ b_r,
    const float* __restrict__ W_z, const float* __restrict__ b_z,
    float* __restrict__ rh, float* __restrict__ zout)
{
    __shared__ __bf16 As[2][BM * BK];
    __shared__ __bf16 Bs[2][BN * BK];

    const int tid  = threadIdx.x;
    const int lane = tid & 31;
    const int wave = tid >> 5;
    const int wm   = wave >> 2;               // 0..1
    const int wn   = wave & 3;                // 0..3
    const int blockN = blockIdx.x * BN;
    const int blockM = blockIdx.y * BM;
    const bool is_z  = (blockIdx.z != 0);

    const float* __restrict__ W    = is_z ? W_z : W_r;
    const float* __restrict__ bias = is_z ? b_z : b_r;

    v8f acc[4][4];
#pragma unroll
    for (int i = 0; i < 4; ++i)
#pragma unroll
        for (int j = 0; j < 4; ++j) acc[i][j] = (v8f)(0.0f);

    // prologue: stage K-tile 0 into buffer 0
    stage_tiles(As[0], Bs[0], x, W, blockM, blockN, 0, tid, true);

    for (int kt = 0; kt < KT; ++kt) {
        __syncthreads();                       // buffer (kt&1) is ready
        const int buf = kt & 1;
        if (kt + 1 < KT) {                     // overlap: fill other buffer
            const int k1 = (kt + 1) * BK;
            const float* asrc = (k1 < ISZ) ? (x + k1) : (h_prev + (k1 - ISZ));
            stage_tiles(As[buf ^ 1], Bs[buf ^ 1], asrc, W, blockM, blockN, k1,
                        tid, kt + 2 < KT);
        }
        mma_step(As[buf], Bs[buf], acc, wm, wn, lane);
    }

    // ---- epilogue: bias + sigmoid (+ r*h_prev fusion) ----
    const int mrow0 = blockM + wm * 64;
    const int ncol0 = blockN + wn * 64;
    const int lhalf = (lane >> 4) * 8;
    const int lcol  = lane & 15;
#pragma unroll
    for (int i = 0; i < 4; ++i) {
#pragma unroll
        for (int j = 0; j < 4; ++j) {
            const int col = ncol0 + j * 16 + lcol;
            const float bv = bias[col];
#pragma unroll
            for (int v = 0; v < 8; ++v) {
                const int m = mrow0 + i * 16 + lhalf + v;
                const size_t idx = (size_t)m * HSZ + col;
                const float s = sigmoidf_(acc[i][j][v] + bv);
                if (is_z) zout[idx] = s;
                else      rh[idx]   = s * h_prev[idx];
            }
        }
    }
}

// =====================  kernel 2: candidate GEMM + full epilogue  ==========
// n = tanh(concat(x, rh) @ W_c^T + b_c); h_new = (1-z)h + z n;
// pot = pot_prev + h_new; gate on (pot - tresh) > 0.
__global__ __launch_bounds__(256) void pglu_gemm_c(
    const float* __restrict__ x,   const float* __restrict__ rh,
    const float* __restrict__ W_c, const float* __restrict__ b_c,
    const float* __restrict__ zbuf, const float* __restrict__ h_prev,
    const float* __restrict__ pot_prev, const float* __restrict__ tresh,
    const float* __restrict__ decay, float* __restrict__ out)
{
    __shared__ __bf16 As[2][BM * BK];
    __shared__ __bf16 Bs[2][BN * BK];

    const int tid  = threadIdx.x;
    const int lane = tid & 31;
    const int wave = tid >> 5;
    const int wm   = wave >> 2;
    const int wn   = wave & 3;
    const int blockN = blockIdx.x * BN;
    const int blockM = blockIdx.y * BM;

    v8f acc[4][4];
#pragma unroll
    for (int i = 0; i < 4; ++i)
#pragma unroll
        for (int j = 0; j < 4; ++j) acc[i][j] = (v8f)(0.0f);

    stage_tiles(As[0], Bs[0], x, W_c, blockM, blockN, 0, tid, true);

    for (int kt = 0; kt < KT; ++kt) {
        __syncthreads();
        const int buf = kt & 1;
        if (kt + 1 < KT) {
            const int k1 = (kt + 1) * BK;
            const float* asrc = (k1 < ISZ) ? (x + k1) : (rh + (k1 - ISZ));
            stage_tiles(As[buf ^ 1], Bs[buf ^ 1], asrc, W_c, blockM, blockN, k1,
                        tid, kt + 2 < KT);
        }
        mma_step(As[buf], Bs[buf], acc, wm, wn, lane);
    }

    const size_t BH = (size_t)BATCH * HSZ;
    const int mrow0 = blockM + wm * 64;
    const int ncol0 = blockN + wn * 64;
    const int lhalf = (lane >> 4) * 8;
    const int lcol  = lane & 15;
#pragma unroll
    for (int i = 0; i < 4; ++i) {
#pragma unroll
        for (int j = 0; j < 4; ++j) {
            const int col = ncol0 + j * 16 + lcol;
            const float bv = b_c[col];
            const float tv = tresh[col];
            const float dv = decay[col];
#pragma unroll
            for (int v = 0; v < 8; ++v) {
                const int m = mrow0 + i * 16 + lhalf + v;
                const size_t idx = (size_t)m * HSZ + col;
                const float n  = tanhf(acc[i][j][v] + bv);
                const float zv = zbuf[idx];
                const float hp = h_prev[idx];
                const float hn = (1.0f - zv) * hp + zv * n;
                const float pot = pot_prev[idx] + hn;
                const float g = pot - tv;                 // relu(g) > 0 <=> g > 0
                out[idx]      = (g > 0.0f) ? pot : 0.0f;            // activated
                out[BH + idx] = (g > 0.0f) ? 0.0f : pot * dv;       // non-gated
            }
        }
    }
}

extern "C" void kernel_launch(void* const* d_in, const int* in_sizes, int n_in,
                              void* d_out, int out_size, void* d_ws, size_t ws_size,
                              hipStream_t stream) {
    (void)in_sizes; (void)n_in; (void)out_size; (void)ws_size;
    const float* x        = (const float*)d_in[0];
    const float* h_prev   = (const float*)d_in[1];
    const float* pot_prev = (const float*)d_in[2];
    const float* W_r      = (const float*)d_in[3];
    const float* b_r      = (const float*)d_in[4];
    const float* W_z      = (const float*)d_in[5];
    const float* b_z      = (const float*)d_in[6];
    const float* W_c      = (const float*)d_in[7];
    const float* b_c      = (const float*)d_in[8];
    const float* tresh    = (const float*)d_in[9];
    const float* decay    = (const float*)d_in[10];
    float* out = (float*)d_out;

    float* rh = (float*)d_ws;                       // sigmoid(r) * h_prev  [B,H]
    float* zb = rh + (size_t)BATCH * HSZ;           // sigmoid(z)           [B,H]

    dim3 block(256, 1, 1);
    pglu_gemm_rz<<<dim3(HSZ / BN, BATCH / BM, 2), block, 0, stream>>>(
        x, h_prev, W_r, b_r, W_z, b_z, rh, zb);
    pglu_gemm_c<<<dim3(HSZ / BN, BATCH / BM, 1), block, 0, stream>>>(
        x, rh, W_c, b_c, zb, h_prev, pot_prev, tresh, decay, out);
}
